// CrystalGNN_27152783245566
// MI455X (gfx1250) — matrix operand order repaired and checked by reading, not compiled, not run on hardware
//
#include <hip/hip_runtime.h>
#include <hip/hip_bf16.h>

// ---------------- types ----------------
typedef __attribute__((ext_vector_type(16))) __bf16 v16bf;
typedef __attribute__((ext_vector_type(8)))  __bf16 v8bf;
typedef __attribute__((ext_vector_type(8)))  float  v8f;

union V16U { v16bf v; v8bf h[2]; };

#define Hh   128
#define ZIN  272
#define ZPAD 288
#define KC   9            // 288 / 32 k-chunks
#define PM   (KC*128*32)  // 36864 packed bf16 elems per weight matrix
#define TILE_E 32         // edges per block tile (two 16-row WMMA M-tiles)
#define SMEM_US (2*PM + TILE_E*ZPAD)   // 82944 ushorts
#define SMEM_BYTES (SMEM_US*2)         // 165888 bytes dynamic LDS

__device__ __forceinline__ unsigned short f2bf(float f) {
    unsigned int u = __float_as_uint(f);
    unsigned int r = (u + 0x7FFFu + ((u >> 16) & 1u)) >> 16;
    return (unsigned short)r;
}

__device__ __forceinline__ void atomAddF(float* p, float v) {
    __hip_atomic_fetch_add(p, v, __ATOMIC_RELAXED, __HIP_MEMORY_SCOPE_AGENT);
}

// ---------------- weight pre-pack: fp32 [L,272,128] -> bf16 B-operand layout ----------------
// packed[matrix][kchunk][col][khalf][i] ; matrix = layer*2 + (0=Wf,1=Ws)
__global__ void prep_weights_kernel(const float* __restrict__ Wf,
                                    const float* __restrict__ Ws,
                                    unsigned short* __restrict__ packed) {
    int idx = blockIdx.x * blockDim.x + threadIdx.x;
    if (idx >= 6 * PM) return;
    int m     = idx / PM;
    int layer = m >> 1, which = m & 1;
    int r     = idx - m * PM;
    int c     = r >> 12;              // kchunk (4096 per chunk)
    int r2    = r & 4095;
    int col   = r2 >> 5;
    int t     = r2 & 31;
    int khalf = t >> 4, i = t & 15;
    int k     = c * 32 + (i < 8 ? khalf * 8 + i : 16 + khalf * 8 + (i - 8));
    const float* W = which ? Ws : Wf;
    float v = (k < ZIN) ? W[((size_t)layer * ZIN + k) * Hh + col] : 0.f;
    packed[idx] = f2bf(v);
}

// ---------------- embedding + input linear: x = [emb(z), xs] @ W0 + b0 ----------------
__global__ void embed_linear_kernel(const int* __restrict__ z,
                                    const float* __restrict__ xs,
                                    const float* __restrict__ emb,
                                    const float* __restrict__ W0,
                                    const float* __restrict__ b0,
                                    float* __restrict__ x, int N) {
    int tid = blockIdx.x * blockDim.x + threadIdx.x;
    if (tid >= N * Hh) return;
    int n = tid >> 7, c = tid & 127;
    int zz = z[n]; zz = zz < 0 ? 0 : (zz > 118 ? 118 : zz);
    const float* ae = emb + (size_t)zz * 64;
    float acc = b0[c];
    #pragma unroll 8
    for (int k = 0; k < 64; ++k) acc = fmaf(ae[k], W0[k * Hh + c], acc);
    const float* xr = xs + (size_t)n * 4;
    #pragma unroll
    for (int k = 0; k < 4; ++k) acc = fmaf(xr[k], W0[(64 + k) * Hh + c], acc);
    x[tid] = acc;
}

// ---------------- generic fill ----------------
__global__ void fill_kernel(float* __restrict__ p, float v, int n) {
    int i = blockIdx.x * blockDim.x + threadIdx.x;
    if (i < n) p[i] = v;
}

// ---------------- edge gated conv via WMMA bf16 ----------------
// block = 256 thr (8 waves). Wave w covers output cols [16w,16w+16).
// Tile = 32 edges (two 16-row M tiles sharing the B loads).
// A(32x288 bf16) staged in LDS, B (both Wf,Ws) resident in LDS.
__global__ __launch_bounds__(256) void edge_gemm_kernel(
    const float* __restrict__ x, const int* __restrict__ ei,
    const float* __restrict__ ea,
    const unsigned short* __restrict__ WfP, const unsigned short* __restrict__ WsP,
    const float* __restrict__ bfv, const float* __restrict__ bsv,
    float* __restrict__ msum, int E, int ntiles) {
    extern __shared__ unsigned short smem[];
    unsigned short* wfl = smem;            // PM
    unsigned short* wsl = smem + PM;       // PM
    unsigned short* at  = smem + 2 * PM;   // TILE_E*ZPAD

    const int tid = threadIdx.x;
    // cooperative weight load (uint4 = 8 ushorts)
    {
        const uint4* sf = (const uint4*)WfP;
        const uint4* ss = (const uint4*)WsP;
        uint4* df = (uint4*)wfl;
        uint4* ds = (uint4*)wsl;
        for (int i = tid; i < PM / 8; i += 256) { df[i] = sf[i]; ds[i] = ss[i]; }
    }
    __syncthreads();

    const int lane  = tid & 31;
    const int wave  = tid >> 5;
    const int colq  = lane & 15;
    const int khalf = lane >> 4;
    const int col   = wave * 16 + colq;
    const float bF  = bfv[col];
    const float bS  = bsv[col];
    const int kb    = khalf * 8;
    const int row0  = colq;        // A row in M-tile 0 held by this lane
    const int row1  = colq + 16;   // A row in M-tile 1

    for (int t = blockIdx.x; t < ntiles; t += gridDim.x) {
        const int base = t * TILE_E;
        // prefetch next tile's edge payload (global_prefetch_b8)
        if (base + TILE_E < E) {
            __builtin_prefetch(ea + (size_t)(base + TILE_E) * 16, 0, 1);
            __builtin_prefetch(ei + E + base + TILE_E, 0, 1);
        }
        // stage A tile: zc = [x[dst] | x[src] | e_ij | pad]
        for (int idx = tid; idx < TILE_E * ZPAD; idx += 256) {
            int el = idx / ZPAD, k = idx - el * ZPAD;
            int e = base + el;
            float v = 0.f;
            if (e < E) {
                if (k < 128)      v = x[(size_t)ei[E + e] * Hh + k];
                else if (k < 256) v = x[(size_t)ei[e] * Hh + (k - 128)];
                else if (k < ZIN) v = ea[(size_t)e * 16 + (k - 256)];
            }
            at[el * ZPAD + k] = f2bf(v);
        }
        __syncthreads();

        v8f accF0 = {0.f,0.f,0.f,0.f,0.f,0.f,0.f,0.f};
        v8f accS0 = {0.f,0.f,0.f,0.f,0.f,0.f,0.f,0.f};
        v8f accF1 = {0.f,0.f,0.f,0.f,0.f,0.f,0.f,0.f};
        v8f accS1 = {0.f,0.f,0.f,0.f,0.f,0.f,0.f,0.f};
        #pragma unroll
        for (int c = 0; c < KC; ++c) {
            V16U a0, a1;
            a0.h[0] = *(const v8bf*)(at + row0 * ZPAD + c * 32 + kb);
            a0.h[1] = *(const v8bf*)(at + row0 * ZPAD + c * 32 + kb + 16);
            a1.h[0] = *(const v8bf*)(at + row1 * ZPAD + c * 32 + kb);
            a1.h[1] = *(const v8bf*)(at + row1 * ZPAD + c * 32 + kb + 16);
            v16bf bF16 = *(const v16bf*)(wfl + ((c * 128 + col) * 2 + khalf) * 16);
            v16bf bS16 = *(const v16bf*)(wsl + ((c * 128 + col) * 2 + khalf) * 16);
            accF0 = __builtin_amdgcn_wmma_f32_16x16x32_bf16(
                        false, a0.v, false, bF16, (short)0, accF0, false, false);
            accF1 = __builtin_amdgcn_wmma_f32_16x16x32_bf16(
                        false, a1.v, false, bF16, (short)0, accF1, false, false);
            accS0 = __builtin_amdgcn_wmma_f32_16x16x32_bf16(
                        false, a0.v, false, bS16, (short)0, accS0, false, false);
            accS1 = __builtin_amdgcn_wmma_f32_16x16x32_bf16(
                        false, a1.v, false, bS16, (short)0, accS1, false, false);
        }

        // epilogue: gate + scatter-add into msum[dst]
        #pragma unroll
        for (int i = 0; i < 8; ++i) {
            int rm = khalf * 8 + i;        // C/D layout: lanes16-31 hold M=8..15
            int e0 = base + rm;            // M-tile 0
            if (e0 < E) {
                int d = ei[E + e0];
                float f = accF0[i] + bF;
                float s = accS0[i] + bS;
                float sig = 1.f / (1.f + __expf(-f));
                float sp  = (s > 20.f) ? s : log1pf(__expf(s));
                atomAddF(&msum[(size_t)d * Hh + col], sig * sp);
            }
            int e1 = base + 16 + rm;       // M-tile 1
            if (e1 < E) {
                int d = ei[E + e1];
                float f = accF1[i] + bF;
                float s = accS1[i] + bS;
                float sig = 1.f / (1.f + __expf(-f));
                float sp  = (s > 20.f) ? s : log1pf(__expf(s));
                atomAddF(&msum[(size_t)d * Hh + col], sig * sp);
            }
        }
        __syncthreads();
    }
}

// ---------------- residual + LayerNorm + ReLU (wave per node, wave32) ----------------
__global__ void ln_relu_kernel(float* __restrict__ x, const float* __restrict__ msum,
                               const float* __restrict__ g, const float* __restrict__ b,
                               int N) {
    int wid  = (blockIdx.x * blockDim.x + threadIdx.x) >> 5;
    int lane = threadIdx.x & 31;
    if (wid >= N) return;
    float4 v = ((const float4*)(x + (size_t)wid * Hh))[lane];
    float4 m = ((const float4*)(msum + (size_t)wid * Hh))[lane];
    v.x += m.x; v.y += m.y; v.z += m.z; v.w += m.w;
    float s = v.x + v.y + v.z + v.w;
    #pragma unroll
    for (int o = 16; o > 0; o >>= 1) s += __shfl_xor(s, o, 32);
    float mu = s * (1.f / 128.f);
    float d0 = v.x - mu, d1 = v.y - mu, d2 = v.z - mu, d3 = v.w - mu;
    float q = d0*d0 + d1*d1 + d2*d2 + d3*d3;
    #pragma unroll
    for (int o = 16; o > 0; o >>= 1) q += __shfl_xor(q, o, 32);
    float inv = rsqrtf(q * (1.f / 128.f) + 1e-5f);
    float4 gg = ((const float4*)g)[lane];
    float4 bb = ((const float4*)b)[lane];
    float4 o4;
    o4.x = fmaxf(d0 * inv * gg.x + bb.x, 0.f);
    o4.y = fmaxf(d1 * inv * gg.y + bb.y, 0.f);
    o4.z = fmaxf(d2 * inv * gg.z + bb.z, 0.f);
    o4.w = fmaxf(d3 * inv * gg.w + bb.w, 0.f);
    ((float4*)(x + (size_t)wid * Hh))[lane] = o4;
}

// ---------------- global mean pool accumulate ----------------
__global__ void pool_accum_kernel(const float* __restrict__ x, const int* __restrict__ batch,
                                  float* __restrict__ sums, float* __restrict__ cnt, int N) {
    int tid = blockIdx.x * blockDim.x + threadIdx.x;
    if (tid >= N * Hh) return;
    int n = tid >> 7, c = tid & 127;
    int gi = batch[n];
    atomAddF(&sums[(size_t)gi * Hh + c], x[tid]);
    if (c == 0) atomAddF(&cnt[gi], 1.f);
}

// ---------------- per-graph MLP head: relu(g@W1+b1)@W2+b2 ----------------
__global__ void head_kernel(const float* __restrict__ sums, const float* __restrict__ cnt,
                            const float* __restrict__ W1, const float* __restrict__ b1,
                            const float* __restrict__ W2, const float* __restrict__ b2,
                            float* __restrict__ out) {
    __shared__ float red[64];
    int g = blockIdx.x, j = threadIdx.x;   // 64 threads
    float c = fmaxf(cnt[g], 1.f);
    float inv = 1.f / c;
    float acc = b1[j];
    #pragma unroll 8
    for (int k = 0; k < Hh; ++k)
        acc = fmaf(sums[(size_t)g * Hh + k] * inv, W1[k * 64 + j], acc);
    red[j] = fmaxf(acc, 0.f) * W2[j];
    __syncthreads();
    for (int s = 32; s > 0; s >>= 1) {
        if (j < s) red[j] += red[j + s];
        __syncthreads();
    }
    if (j == 0) out[g] = red[0] + b2[0];
}

// ---------------- launch ----------------
extern "C" void kernel_launch(void* const* d_in, const int* in_sizes, int n_in,
                              void* d_out, int out_size, void* d_ws, size_t ws_size,
                              hipStream_t stream) {
    const int* z        = (const int*)d_in[0];
    const float* xs     = (const float*)d_in[1];
    const int* ei       = (const int*)d_in[2];
    const float* ea     = (const float*)d_in[3];
    const int* batch    = (const int*)d_in[4];
    const float* emb    = (const float*)d_in[5];
    const float* W0     = (const float*)d_in[6];
    const float* b0     = (const float*)d_in[7];
    const float* Wf     = (const float*)d_in[8];
    const float* bfv    = (const float*)d_in[9];
    const float* Ws     = (const float*)d_in[10];
    const float* bsv    = (const float*)d_in[11];
    const float* lng    = (const float*)d_in[12];
    const float* lnb    = (const float*)d_in[13];
    const float* W1     = (const float*)d_in[14];
    const float* b1     = (const float*)d_in[15];
    const float* W2     = (const float*)d_in[16];
    const float* b2     = (const float*)d_in[17];
    float* out          = (float*)d_out;

    const int N = in_sizes[0];
    const int E = in_sizes[2] / 2;
    const int G = 256;
    const int ntiles = (E + TILE_E - 1) / TILE_E;

    // workspace carve-up (all 16B aligned by construction)
    float* xbuf = (float*)d_ws;
    float* msum = xbuf + (size_t)N * Hh;
    float* sums = msum + (size_t)N * Hh;
    float* cnt  = sums + (size_t)G * Hh;
    unsigned short* packed = (unsigned short*)(cnt + G);

    (void)hipFuncSetAttribute((const void*)edge_gemm_kernel,
                              hipFuncAttributeMaxDynamicSharedMemorySize, SMEM_BYTES);

    // pre-pack weights to bf16 B-operand layout
    prep_weights_kernel<<<(6 * PM + 255) / 256, 256, 0, stream>>>(Wf, Ws, packed);

    // input projection
    embed_linear_kernel<<<(N * Hh + 255) / 256, 256, 0, stream>>>(z, xs, emb, W0, b0, xbuf, N);

    int gemmBlocks = ntiles < 2048 ? ntiles : 2048;
    for (int l = 0; l < 3; ++l) {
        fill_kernel<<<(N * Hh + 255) / 256, 256, 0, stream>>>(msum, 0.f, N * Hh);
        edge_gemm_kernel<<<gemmBlocks, 256, SMEM_BYTES, stream>>>(
            xbuf, ei, ea,
            packed + (size_t)(2 * l) * PM, packed + (size_t)(2 * l + 1) * PM,
            bfv + l * Hh, bsv + l * Hh, msum, E, ntiles);
        ln_relu_kernel<<<(N * 32 + 255) / 256, 256, 0, stream>>>(
            xbuf, msum, lng + l * Hh, lnb + l * Hh, N);
    }

    // pooling + head
    fill_kernel<<<(G * Hh + G + 255) / 256, 256, 0, stream>>>(sums, 0.f, G * Hh + G);
    pool_accum_kernel<<<(N * Hh + 255) / 256, 256, 0, stream>>>(xbuf, batch, sums, cnt, N);
    head_kernel<<<G, 64, 0, stream>>>(sums, cnt, W1, b1, W2, b2, out);
}
